// FeedForwardQuantum_65481071409560
// MI455X (gfx1250) — matrix operand order repaired and checked by reading, not compiled, not run on hardware
//
#include <hip/hip_runtime.h>
#include <hip/hip_bf16.h>

// ---------------------------------------------------------------------------
// FeedForwardQuantum for MI455X (gfx1250, wave32, WMMA + async-to-LDS)
//
// Math: the CNOT-ring RX circuit collapses to
//   qout[j] = prod_{i=0..j} cos(x_i + p_i)   (j >= 1)
//   qout[0] = prod_{i=1..9} cos(x_i + p_i)
// then  out = relu(qout @ W1^T) @ W2^T.
// GEMM2 (16384 x 1024 x 4096 = 137 GFLOP) dominates -> f16 WMMA w/ f32 acc.
// ---------------------------------------------------------------------------

typedef _Float16 half8  __attribute__((ext_vector_type(8)));
typedef _Float16 half16 __attribute__((ext_vector_type(16)));
typedef float    float8 __attribute__((ext_vector_type(8)));
typedef int      int4v  __attribute__((ext_vector_type(4)));

// Global-address-space views: force global_load/global_store (LOADcnt only).
typedef __attribute__((address_space(1))) const _Float16 gc_half;
typedef __attribute__((address_space(1))) const half8   gc_half8;
typedef __attribute__((address_space(1))) float         g_float;
typedef __attribute__((address_space(1))) int4v         g_int4;
// LDS views: force ds_load.
typedef __attribute__((address_space(3))) _Float16       l_half;
typedef __attribute__((address_space(3))) const half8   lc_half8;
typedef __attribute__((address_space(3))) int4v         l_int4;

#define NQ      10
#define M_ROWS  16384      // B*S = 32*512
#define F_DIM   4096
#define E_DIM   1024

#define BROW_PAD 264       // 32-row B panel, row stride 264 halfs (528B):
                           // dword stride 132 == 4 (mod 64) -> conflict-free

// ---- async copy global(16B) -> LDS, tracked by ASYNCcnt -------------------
// Builtin signature (from toolchain): (v4i AS1* src, v4i AS3* dst, Ii, Ii)
__device__ __forceinline__ void async_ld_b128(gc_half* g, l_half* l) {
#if __has_builtin(__builtin_amdgcn_global_load_async_to_lds_b128)
  __builtin_amdgcn_global_load_async_to_lds_b128((g_int4*)g, (l_int4*)l, 0, 0);
#else
  unsigned lds_off = (unsigned)reinterpret_cast<unsigned long long>(l);
  unsigned long long ga = reinterpret_cast<unsigned long long>(g);
  asm volatile("global_load_async_to_lds_b128 %0, %1, off"
               :: "v"(lds_off), "v"(ga) : "memory");
#endif
}

__device__ __forceinline__ void wait_async0() {
#if __has_builtin(__builtin_amdgcn_s_wait_asynccnt)
  __builtin_amdgcn_s_wait_asynccnt(0);
#else
  asm volatile("s_wait_asynccnt 0x0" ::: "memory");
#endif
}

// ---------------------------------------------------------------------------
// Kernel 1: closed-form circuit expectation values. One thread per row.
// ---------------------------------------------------------------------------
__global__ __launch_bounds__(256)
void qexp_kernel(const float* __restrict__ x,
                 const float* __restrict__ qp,
                 float* __restrict__ qout) {
  const int r = blockIdx.x * 256 + threadIdx.x;     // grid sized exactly
  float c[NQ];
#pragma unroll
  for (int i = 0; i < NQ; ++i) c[i] = cosf(x[r * NQ + i] + qp[i]);
  float o[NQ];
  float pre = 1.0f;
#pragma unroll
  for (int i = 0; i < NQ; ++i) { pre *= c[i]; o[i] = pre; }
  float s = 1.0f;
#pragma unroll
  for (int i = 1; i < NQ; ++i) s *= c[i];
  o[0] = s;                                         // ring CNOT closes the loop
#pragma unroll
  for (int i = 0; i < NQ; ++i) qout[r * NQ + i] = o[i];
}

// ---------------------------------------------------------------------------
// Kernel 2: transpose + f32->f16 W2 (E x F) -> W2T (F x E, N contiguous).
// ---------------------------------------------------------------------------
__global__ __launch_bounds__(256)
void w2t_kernel(const float* __restrict__ W2, _Float16* __restrict__ BT) {
  __shared__ float tile[32][33];                    // +1 pad: no bank conflicts
  const int k0 = blockIdx.x * 32;
  const int n0 = blockIdx.y * 32;
  const int tx = threadIdx.x;                       // 0..31
  const int ty = threadIdx.y;                       // 0..7
#pragma unroll
  for (int j = ty; j < 32; j += 8)
    tile[j][tx] = W2[(size_t)(n0 + j) * F_DIM + (k0 + tx)];
  __syncthreads();
#pragma unroll
  for (int j = ty; j < 32; j += 8)
    BT[(size_t)(k0 + j) * E_DIM + (n0 + tx)] = (_Float16)tile[tx][j];
}

// ---------------------------------------------------------------------------
// Kernel 3: H = relu(qout @ W1^T), stored f16.
// ---------------------------------------------------------------------------
__global__ __launch_bounds__(256)
void hgen_kernel(const float* __restrict__ qout,
                 const float* __restrict__ W1,
                 _Float16* __restrict__ H) {
  const int r = blockIdx.x;                         // 0..16383
  const int f = blockIdx.y * 256 + threadIdx.x;     // 0..4095
  const float* q = qout + r * NQ;
  const float* w = W1 + f * NQ;
  float s = 0.0f;
#pragma unroll
  for (int i = 0; i < NQ; ++i) s = fmaf(q[i], w[i], s);
  s = fmaxf(s, 0.0f);
  H[(size_t)r * F_DIM + f] = (_Float16)s;
}

// ---------------------------------------------------------------------------
// Kernel 4: C = H(16384x4096 f16) * W2T(4096x1024 f16).
// Block = 256 threads = 8 waves (2 along M x 4 along N); block tile 128x256;
// wave tile 64x64 = 4x4 v_wmma_f32_16x16x32_f16 tiles; K-step 32.
//
// B panels (32 x 256 f16, 16KB) are staged into LDS once per block with
// GLOBAL_LOAD_ASYNC_TO_LDS_B128 (ASYNCcnt), double-buffered so the DMA of
// step k+32 overlaps the WMMA of step k. A fragments load straight from
// global (L2-resident H). One barrier + one s_wait_asynccnt per K-step.
//
// Fragment layouts per CDNA5 ISA 7.12.2 (16-bit, wave32):
//  A (16x32 MxK): lanes 0-15: V0-3=K0..7, V4-7=K16..23;
//                 lanes 16-31: V0-3=K8..15, V4-7=K24..31.
//  B (32x16 KxN): N = lane&15; K per lane half / VGPR same as A.
//  C/D (16x16 f32): row = v + 8*(lane>>4), col = lane&15.
// ---------------------------------------------------------------------------
__global__ __launch_bounds__(256)
void gemm2_kernel(const _Float16* __restrict__ Hm,
                  const _Float16* __restrict__ BT,
                  float* __restrict__ C) {
  __shared__ __attribute__((aligned(16))) _Float16 sB[2][32 * BROW_PAD];

  const int tid  = threadIdx.x;
  const int lane = tid & 31;
  const int wave = tid >> 5;                         // 0..7
  const int m0   = blockIdx.y * 128 + (wave >> 2) * 64;
  const int nblk = blockIdx.x * 256;                 // block's N origin
  const int nloc = (wave & 3) * 64;                  // wave's N within block

  const int laneM = lane & 15;
  const int kSel  = (lane >> 4) << 3;                // 0 or 8

  float8 acc[4][4];
#pragma unroll
  for (int mt = 0; mt < 4; ++mt)
#pragma unroll
    for (int nt = 0; nt < 4; ++nt)
      acc[mt][nt] = (float8)0.0f;

  gc_half* aptr[4];
#pragma unroll
  for (int mt = 0; mt < 4; ++mt)
    aptr[mt] = (gc_half*)(Hm + (size_t)(m0 + mt * 16 + laneM) * F_DIM + kSel);

  // ---- stage B panel rows kk..kk+31 into sB[buf]: 256 thr x 4 x 16B ------
  auto stageB = [&](int buf, int kk) {
#pragma unroll
    for (int j = 0; j < 4; ++j) {
      const int c    = j * 256 + tid;                // 0..1023 16B-chunks
      const int row  = c >> 5;                       // 0..31
      const int col8 = c & 31;                       // 16B chunk in row
      gc_half* g = (gc_half*)(BT + (size_t)(kk + row) * E_DIM + nblk + col8 * 8);
      l_half*  l = (l_half*)&sB[buf][row * BROW_PAD + col8 * 8];
      async_ld_b128(g, l);
    }
  };

  stageB(0, 0);
  wait_async0();
  __syncthreads();

  int cur = 0;
#pragma unroll 1
  for (int kk = 0; kk < F_DIM; kk += 32) {
    if (kk + 32 < F_DIM) stageB(cur ^ 1, kk + 32);   // overlap next DMA

    half16 a[4];
#pragma unroll
    for (int mt = 0; mt < 4; ++mt) {
      half8 lo = *(gc_half8*)(aptr[mt]);             // K kbase..kbase+7
      half8 hi = *(gc_half8*)(aptr[mt] + 16);        // K kbase+16..kbase+23
      a[mt] = __builtin_shufflevector(lo, hi, 0, 1, 2, 3, 4, 5, 6, 7,
                                              8, 9, 10, 11, 12, 13, 14, 15);
      aptr[mt] += 32;
    }
    half16 b[4];
#pragma unroll
    for (int nt = 0; nt < 4; ++nt) {
      const _Float16* p = &sB[cur][nloc + nt * 16 + laneM];
#pragma unroll
      for (int s = 0; s < 16; ++s)
        b[nt][s] = p[(kSel + (s & 7) + ((s >> 3) << 4)) * BROW_PAD];
    }
#pragma unroll
    for (int mt = 0; mt < 4; ++mt)
#pragma unroll
      for (int nt = 0; nt < 4; ++nt)
        acc[mt][nt] = __builtin_amdgcn_wmma_f32_16x16x32_f16(
            /*neg_a=*/false, a[mt], /*neg_b=*/false, b[nt],
            /*c_mod=*/(short)0, acc[mt][nt],
            /*reuse_a=*/false, /*reuse_b=*/false);

    wait_async0();        // next panel landed (overlapped with WMMA above)
    __syncthreads();      // also fences buffer reuse (read in iter k-32)
    cur ^= 1;
  }

  // D layout: row = v + 8*(lane>>4), col = lane&15
  g_float* Cg = (g_float*)C;
  const int rowHalf = (lane >> 4) << 3;
  const int col     = lane & 15;
#pragma unroll
  for (int mt = 0; mt < 4; ++mt)
#pragma unroll
    for (int nt = 0; nt < 4; ++nt)
#pragma unroll
      for (int v = 0; v < 8; ++v)
        Cg[(size_t)(m0 + mt * 16 + rowHalf + v) * E_DIM + (nblk + nloc + nt * 16 + col)] =
            acc[mt][nt][v];
}

// ---------------------------------------------------------------------------
// Launch. Workspace (~144 MB):
//   [0, 640KB)      qout  (f32 16384x10)
//   [1MB, 129MB)    H     (f16 16384x4096)
//   [129MB, 137MB)  W2T   (f16 4096x1024)
// ---------------------------------------------------------------------------
extern "C" void kernel_launch(void* const* d_in, const int* in_sizes, int n_in,
                              void* d_out, int out_size, void* d_ws, size_t ws_size,
                              hipStream_t stream) {
  (void)in_sizes; (void)n_in; (void)out_size; (void)ws_size;
  const float* x  = (const float*)d_in[0];
  const float* qp = (const float*)d_in[1];
  const float* W1 = (const float*)d_in[2];
  const float* W2 = (const float*)d_in[3];
  float* out = (float*)d_out;

  char* ws = (char*)d_ws;
  float*    qout = (float*)ws;
  _Float16* Hbuf = (_Float16*)(ws + (size_t)(1u << 20));
  _Float16* BT   = (_Float16*)(ws + (size_t)(1u << 20) + (size_t)134217728u);

  // 1) closed-form circuit
  qexp_kernel<<<dim3(M_ROWS / 256), dim3(256), 0, stream>>>(x, qp, qout);
  // 2) W2 -> K-major f16
  w2t_kernel<<<dim3(F_DIM / 32, E_DIM / 32), dim3(32, 8), 0, stream>>>(W2, BT);
  // 3) H = relu(qout @ W1^T) in f16
  hgen_kernel<<<dim3(M_ROWS, F_DIM / 256), dim3(256), 0, stream>>>(qout, W1, Hbuf);
  // 4) out = H @ W2^T via WMMA + async-LDS double buffering
  gemm2_kernel<<<dim3(E_DIM / 256, M_ROWS / 128), dim3(256), 0, stream>>>(Hbuf, BT, out);
}
